// CrossAttentionLayer_47193100648749
// MI455X (gfx1250) — matrix operand order repaired and checked by reading, not compiled, not run on hardware
//
#include <hip/hip_runtime.h>

// ---------------------------------------------------------------------------
// Problem constants (from reference)
// ---------------------------------------------------------------------------
#define BB   128
#define LT   96
#define LI   64
#define DT   1024
#define DI   2048
#define DD   1024
#define HH   4
#define DH   256   // DD / HH

// ---------------------------------------------------------------------------
// CDNA5 async global->LDS path (guarded; falls back to synchronous staging)
// ---------------------------------------------------------------------------
#if defined(__has_builtin)
#  if __has_builtin(__builtin_amdgcn_global_load_async_to_lds_b128)
#    define USE_ASYNC_LDS 1
#  endif
#endif
#ifndef USE_ASYNC_LDS
#  define USE_ASYNC_LDS 0
#endif

typedef __attribute__((__vector_size__(16))) int i32x4;   // builtin ptr elem type

#if USE_ASYNC_LDS
__device__ __forceinline__ void wait_asynccnt0() {
#if defined(__has_builtin) && __has_builtin(__builtin_amdgcn_s_wait_asynccnt)
    __builtin_amdgcn_s_wait_asynccnt(0);
#else
    asm volatile("s_wait_asynccnt 0x0" ::: "memory");
#endif
}
#endif

// ---------------------------------------------------------------------------
// Vector types for WMMA
// ---------------------------------------------------------------------------
typedef __attribute__((ext_vector_type(16))) __bf16          bf16x16;
typedef __attribute__((ext_vector_type(8)))  float           f32x8;
typedef __attribute__((ext_vector_type(8)))  unsigned short  u16x8;
typedef __attribute__((ext_vector_type(16))) unsigned short  u16x16;

__device__ __forceinline__ unsigned short f2bf(float f) {
    union { float f; unsigned int u; } c; c.f = f;
    unsigned int u = c.u;
    u += 0x7FFFu + ((u >> 16) & 1u);       // round-to-nearest-even
    return (unsigned short)(u >> 16);
}
__device__ __forceinline__ float bf2f(unsigned short h) {
    union { unsigned int u; float f; } c; c.u = ((unsigned int)h) << 16;
    return c.f;
}

// A-fragment: 16x32 bf16 tile, matrix row-major with leading dim `ld` (elems).
// 16-bit A layout (matches B): lanes 0-15: K = k0+0..15 ; lanes 16-31: K+16.
__device__ __forceinline__ bf16x16 load_a_frag(const unsigned short* base, int ld,
                                               int k0, int lane) {
    int m  = lane & 15;
    int kb = (lane >> 4) << 4;             // 0 or 16
    const unsigned short* p = base + (size_t)m * ld + k0 + kb;
    u16x16 c = *(const u16x16*)(p);
    union { u16x16 u; bf16x16 b; } cv; cv.u = c; return cv.b;
}

// B-fragment: 32x16 bf16 tile from a K-contiguous (transposed) matrix Bt[N][K].
// B layout used here (contiguous k, matches A): lanes 0-15: K = k0+0..15 at N=lane; lanes 16-31: K+16.
__device__ __forceinline__ bf16x16 load_b_frag(const unsigned short* baseN0, int ld,
                                               int k0, int lane) {
    int n  = lane & 15;
    int ko = (lane >> 4) << 4;             // 0 or 16
    const unsigned short* p = baseN0 + (size_t)n * ld + k0 + ko;
    union { u16x16 u; bf16x16 b; } cv; cv.u = *(const u16x16*)p; return cv.b;
}

__device__ __forceinline__ f32x8 wmma_bf16(bf16x16 a, bf16x16 b, f32x8 c) {
    return __builtin_amdgcn_wmma_f32_16x16x32_bf16(false, a, false, b,
                                                   (short)0, c, false, false);
}

// ---------------------------------------------------------------------------
// Elementwise fp32 -> bf16
// ---------------------------------------------------------------------------
__global__ void __launch_bounds__(256) cvt_bf16_kernel(const float* __restrict__ in,
                                                       unsigned short* __restrict__ out,
                                                       int n) {
    int i = blockIdx.x * 256 + threadIdx.x;
    if (i < n) out[i] = f2bf(in[i]);
}

// W fp32 [K,N] row-major  ->  Wt bf16 [N,K] row-major
__global__ void __launch_bounds__(256) wcvt_kernel(const float* __restrict__ W,
                                                   unsigned short* __restrict__ Wt,
                                                   int K, int N) {
    int idx = blockIdx.x * 256 + threadIdx.x;
    if (idx >= K * N) return;
    int n = idx / K, k = idx - n * K;
    Wt[idx] = f2bf(W[(size_t)k * N + n]);
}

// V bf16 [B*L, DD] -> Vt bf16 [B][H][DH][L]
__global__ void __launch_bounds__(256) vtrans_kernel(const unsigned short* __restrict__ V,
                                                     unsigned short* __restrict__ Vt,
                                                     int L) {
    int idx = blockIdx.x * 256 + threadIdx.x;
    int total = BB * HH * DH * L;
    if (idx >= total) return;
    int t = idx;
    int k = t % L;  t /= L;
    int d = t % DH; t /= DH;
    int h = t % HH; t /= HH;
    int b = t;
    Vt[idx] = V[((size_t)(b * L + k)) * DD + h * DH + d];
}

// ---------------------------------------------------------------------------
// GEMM: C[M,N] = A[M,K] @ Bt[N,K]^T (+ bias), all bf16 except fp32 bias.
// Block = 256 threads (8 waves), block tile 32(M) x 256(N).
// A staged in LDS; on CDNA5 via double-buffered async global->LDS copies
// (ASYNCcnt path), overlapping the copy of chunk i+1 with WMMAs on chunk i.
// ---------------------------------------------------------------------------
__global__ void __launch_bounds__(256) gemm_bf16_kernel(
        const unsigned short* __restrict__ A,
        const unsigned short* __restrict__ Bt,
        const float* __restrict__ bias,
        unsigned short* __restrict__ C,
        int M, int K, int N) {
    __shared__ __align__(32) unsigned short As[2][32][128];

    int tid  = threadIdx.x;
    int lane = tid & 31;
    int w    = tid >> 5;
    int wm   = w >> 2;                 // 0..1
    int wn   = w & 3;                  // 0..3
    int m0   = blockIdx.x * 32 + wm * 16;
    int n0   = blockIdx.y * 256 + wn * 64;

    f32x8 acc[4];
#pragma unroll
    for (int j = 0; j < 4; ++j) acc[j] = (f32x8){0.f,0.f,0.f,0.f,0.f,0.f,0.f,0.f};

    int lrow = tid >> 3;               // 0..31
    int lseg = tid & 7;                // 0..7
    const unsigned short* arow = A + (size_t)(blockIdx.x * 32 + lrow) * K;

    // 128 WMMA-K per LDS chunk, computed from buffer `buf`
    auto compute_chunk = [&](const unsigned short* asb, int k0) {
#pragma unroll
        for (int kk = 0; kk < 128; kk += 32) {
            bf16x16 a = load_a_frag(asb, 128, kk, lane);
#pragma unroll
            for (int j = 0; j < 4; ++j) {
                bf16x16 b = load_b_frag(Bt + (size_t)(n0 + j * 16) * K,
                                        K, k0 + kk, lane);
                acc[j] = wmma_bf16(a, b, acc[j]);
            }
        }
    };

#if USE_ASYNC_LDS
    // Each thread copies 32B of its chunk row as two async b128 transfers.
    auto issue_copy = [&](int k0, int buf) {
        const unsigned short* src = &arow[k0 + lseg * 16];
        unsigned short*       dst = &As[buf][lrow][lseg * 16];
        __builtin_amdgcn_global_load_async_to_lds_b128(
            (i32x4*)(src),     (i32x4*)(dst),     0, 0);
        __builtin_amdgcn_global_load_async_to_lds_b128(
            (i32x4*)(src + 8), (i32x4*)(dst + 8), 0, 0);
    };

    issue_copy(0, 0);
    int it = 0;
    for (int k0 = 0; k0 < K; k0 += 128, ++it) {
        int buf = it & 1;
        wait_asynccnt0();              // this wave's copy of chunk `it` landed
        __syncthreads();               // all waves' copies visible; prior reads done
        if (k0 + 128 < K)
            issue_copy(k0 + 128, buf ^ 1);   // overlap next copy with compute
        compute_chunk(&As[buf][wm * 16][0], k0);
    }
#else
    for (int k0 = 0; k0 < K; k0 += 128) {
        __syncthreads();
        *(u16x16*)&As[0][lrow][lseg * 16] = *(const u16x16*)&arow[k0 + lseg * 16];
        if (k0 + 128 < K)
            __builtin_prefetch(&arow[k0 + 128 + lseg * 16], 0, 1);
        __syncthreads();
        compute_chunk(&As[0][wm * 16][0], k0);
    }
#endif

    int nlo  = lane & 15;
    int half = (lane >> 4) << 3;       // 0 or 8
#pragma unroll
    for (int j = 0; j < 4; ++j) {
        int col = n0 + j * 16 + nlo;
        float bv = bias ? bias[col] : 0.f;
#pragma unroll
        for (int r = 0; r < 8; ++r) {
            int row = m0 + r + half;
            C[(size_t)row * N + col] = f2bf(acc[j][r] + bv);
        }
    }
}

// ---------------------------------------------------------------------------
// Fused attention per (batch, head): scores = QK^T/16, mask(NEG), softmax,
// mask-zero, out = attn @ V.  Block = 128 threads (4 waves); each wave owns
// one 16-query strip.  Qm/Km: bf16 [B*L, DD]; Vt: bf16 [B][H][DH][Lk].
// ---------------------------------------------------------------------------
__global__ void __launch_bounds__(128) attn_kernel(
        const unsigned short* __restrict__ Qm,
        const unsigned short* __restrict__ Km,
        const unsigned short* __restrict__ Vt,
        const int* __restrict__ qmask,
        const int* __restrict__ kmask,
        unsigned short* __restrict__ Ctx,
        int Lq, int Lk) {
    __shared__ __align__(32) float          sc[4][16][96];
    __shared__ __align__(32) unsigned short at[4][16][96];

    int tid  = threadIdx.x;
    int lane = tid & 31;
    int w    = tid >> 5;
    int bh   = blockIdx.x;
    int b    = bh >> 2;
    int h    = bh & 3;

    const unsigned short* qb  = Qm + (size_t)b * Lq * DD + h * DH;
    const unsigned short* kb  = Km + (size_t)b * Lk * DD + h * DH;
    const unsigned short* vtb = Vt + (size_t)bh * DH * Lk;
    const int* qm = qmask + (size_t)b * Lq;
    const int* km = kmask + (size_t)b * Lk;

    int nStrips = Lq >> 4;
    int iters   = (nStrips + 3) >> 2;
    int nlo     = lane & 15;
    int half    = (lane >> 4) << 3;

    for (int it = 0; it < iters; ++it) {
        int  s      = it * 4 + w;
        bool active = (s < nStrips);

        if (active) {                                   // wave-uniform
            const unsigned short* aBase = qb + (size_t)(s * 16) * DD;
            for (int nt = 0; nt < (Lk >> 4); ++nt) {
                f32x8 acc = (f32x8){0.f,0.f,0.f,0.f,0.f,0.f,0.f,0.f};
                const unsigned short* bBase = kb + (size_t)(nt * 16) * DD;
#pragma unroll
                for (int kk = 0; kk < DH; kk += 32) {
                    bf16x16 a = load_a_frag(aBase, DD, kk, lane);
                    bf16x16 bmat = load_b_frag(bBase, DD, kk, lane);
                    acc = wmma_bf16(a, bmat, acc);
                }
#pragma unroll
                for (int r = 0; r < 8; ++r)
                    sc[w][r + half][nt * 16 + nlo] = acc[r] * 0.0625f;  // 1/sqrt(256)
            }
        }
        __syncthreads();

        if (active && lane < 16) {                       // softmax, row = lane
            int   qrow = s * 16 + lane;
            int   qok  = qm[qrow];
            float mx   = -3.0e38f;
            for (int k2 = 0; k2 < Lk; ++k2) {
                float v = (km[k2] != 0) ? sc[w][lane][k2] : -2147483648.0f;
                sc[w][lane][k2] = v;
                mx = fmaxf(mx, v);
            }
            float sum = 0.f;
            for (int k2 = 0; k2 < Lk; ++k2) {
                float e = (km[k2] != 0) ? __expf(sc[w][lane][k2] - mx) : 0.f;
                sc[w][lane][k2] = e;
                sum += e;
            }
            float inv = (qok != 0 && sum > 0.f) ? (1.f / sum) : 0.f;
            for (int k2 = 0; k2 < Lk; ++k2)
                at[w][lane][k2] = f2bf(sc[w][lane][k2] * inv);
        }
        __syncthreads();

        if (active) {                                    // out = attn @ V
            const unsigned short* aBase = &at[w][0][0];  // lda = 96 (padded)
            for (int nt2 = 0; nt2 < DH / 16; ++nt2) {
                f32x8 acc = (f32x8){0.f,0.f,0.f,0.f,0.f,0.f,0.f,0.f};
                const unsigned short* bBase = vtb + (size_t)(nt2 * 16) * Lk;
                for (int kk = 0; kk < Lk; kk += 32) {
                    bf16x16 a = load_a_frag(aBase, 96, kk, lane);
                    bf16x16 bmat = load_b_frag(bBase, Lk, kk, lane);
                    acc = wmma_bf16(a, bmat, acc);
                }
#pragma unroll
                for (int r = 0; r < 8; ++r) {
                    int row = b * Lq + s * 16 + r + half;
                    Ctx[(size_t)row * DD + h * DH + nt2 * 16 + nlo] = f2bf(acc[r]);
                }
            }
        }
        __syncthreads();
    }
}

// ---------------------------------------------------------------------------
// LayerNorm over D=1024, in-place on bf16; optional fp32 copy to fout.
// ---------------------------------------------------------------------------
__global__ void __launch_bounds__(256) ln_kernel(unsigned short* __restrict__ X,
                                                 const float* __restrict__ g,
                                                 const float* __restrict__ bta,
                                                 float* __restrict__ fout) {
    __shared__ float red[256];
    int row = blockIdx.x, tid = threadIdx.x;
    unsigned short* xp = X + (size_t)row * DD;

    float v[4]; float s = 0.f;
#pragma unroll
    for (int k = 0; k < 4; ++k) { v[k] = bf2f(xp[tid + k * 256]); s += v[k]; }
    red[tid] = s; __syncthreads();
    for (int off = 128; off > 0; off >>= 1) {
        if (tid < off) red[tid] += red[tid + off];
        __syncthreads();
    }
    float mean = red[0] * (1.f / DD);
    __syncthreads();

    float s2 = 0.f;
#pragma unroll
    for (int k = 0; k < 4; ++k) { float d = v[k] - mean; s2 += d * d; }
    red[tid] = s2; __syncthreads();
    for (int off = 128; off > 0; off >>= 1) {
        if (tid < off) red[tid] += red[tid + off];
        __syncthreads();
    }
    float var  = red[0] * (1.f / DD);
    float rstd = rsqrtf(var + 1e-3f);

#pragma unroll
    for (int k = 0; k < 4; ++k) {
        int c = tid + k * 256;
        float y = (v[k] - mean) * rstd * g[c] + bta[c];
        xp[c] = f2bf(y);
        if (fout) fout[(size_t)row * DD + c] = y;
    }
}

// ---------------------------------------------------------------------------
// Orchestration
// ---------------------------------------------------------------------------
static inline void gemm(const unsigned short* A, const unsigned short* Bt,
                        const float* bias, unsigned short* C,
                        int M, int K, int N, hipStream_t s) {
    dim3 grid(M / 32, N / 256);
    gemm_bf16_kernel<<<grid, 256, 0, s>>>(A, Bt, bias, C, M, K, N);
}

extern "C" void kernel_launch(void* const* d_in, const int* in_sizes, int n_in,
                              void* d_out, int out_size, void* d_ws, size_t ws_size,
                              hipStream_t stream) {
    (void)in_sizes; (void)n_in; (void)out_size; (void)ws_size;

    const float* te    = (const float*)d_in[0];
    const float* ie    = (const float*)d_in[1];
    const int*   tmask = (const int*)d_in[2];
    const int*   imask = (const int*)d_in[3];
    const float* W_tp  = (const float*)d_in[4];
    const float* b_tp  = (const float*)d_in[5];
    const float* W_ip  = (const float*)d_in[6];
    const float* b_ip  = (const float*)d_in[7];
    const float* Wraw[16];                       // 8..23 in order
    for (int i = 0; i < 16; ++i) Wraw[i] = (const float*)d_in[8 + i];
    const float* ln_g[4] = { (const float*)d_in[24], (const float*)d_in[26],
                             (const float*)d_in[28], (const float*)d_in[30] };
    const float* ln_b[4] = { (const float*)d_in[25], (const float*)d_in[27],
                             (const float*)d_in[29], (const float*)d_in[31] };

    const int MT = BB * LT;                      // 12288
    const int MI = BB * LI;                      // 8192
    const size_t S_TE = (size_t)BB * LT * DT;    // 12.6M
    const size_t S_IE = (size_t)BB * LI * DI;    // 16.8M
    const size_t S_TP = (size_t)MT * DD;
    const size_t S_IP = (size_t)MI * DD;

    unsigned short* ws = (unsigned short*)d_ws;
    size_t off = 0;
    unsigned short* te_bf = ws + off; off += S_TE;
    unsigned short* ie_bf = ws + off; off += S_IE;
    unsigned short* tp_bf = ws + off; off += S_TP;
    unsigned short* ip_bf = ws + off; off += S_IP;
    unsigned short* ta_bf = ws + off; off += S_IP;   // text branch  [B,LI,D]
    unsigned short* ia_bf = ws + off; off += S_TP;   // image branch [B,LT,D]
    unsigned short* Qb    = ws + off; off += S_TP;
    unsigned short* Kb    = ws + off; off += S_TP;
    unsigned short* Vb    = ws + off; off += S_TP;
    unsigned short* Vt    = ws + off; off += S_TP;
    unsigned short* Cx    = ws + off; off += S_TP;
    unsigned short* Wt    = ws + off;                // weights, 20.97M elems

    // weight slot offsets in Wt (elems). order: tp, ip, then Wraw[0..15]
    const size_t M1 = (size_t)1048576, M2 = (size_t)2097152;
    size_t woff[18]; size_t wsz[18];
    {
        size_t o = 0;
        woff[0] = o; wsz[0] = M1; o += M1;           // W_tp  [1024x1024]
        woff[1] = o; wsz[1] = M2; o += M2;           // W_ip  [2048x1024]
        for (int i = 0; i < 16; ++i) {               // ta(q,k,v,o) ia(...) ts is
            size_t sz = (i == 6) ? M2 : M1;          // ia_Wv has K = DI = 2048
            woff[2 + i] = o; wsz[2 + i] = sz; o += sz;
        }
    }

    // ---- conversions ----
    cvt_bf16_kernel<<<(int)(S_TE / 256), 256, 0, stream>>>(te, te_bf, (int)S_TE);
    cvt_bf16_kernel<<<(int)(S_IE / 256), 256, 0, stream>>>(ie, ie_bf, (int)S_IE);
    wcvt_kernel<<<(int)(M1 / 256), 256, 0, stream>>>(W_tp, Wt + woff[0], DT, DD);
    wcvt_kernel<<<(int)(M2 / 256), 256, 0, stream>>>(W_ip, Wt + woff[1], DI, DD);
    for (int i = 0; i < 16; ++i) {
        int K = (i == 2) ? DT : (i == 6) ? DI : DD;  // ta_Wv: K=DT; ia_Wv: K=DI
        wcvt_kernel<<<(int)(wsz[2 + i] / 256), 256, 0, stream>>>(
            Wraw[i], Wt + woff[2 + i], K, DD);
    }
    const unsigned short* w_tp  = Wt + woff[0];
    const unsigned short* w_ip  = Wt + woff[1];
    const unsigned short* w_taq = Wt + woff[2],  *w_tak = Wt + woff[3];
    const unsigned short* w_tav = Wt + woff[4],  *w_tao = Wt + woff[5];
    const unsigned short* w_iaq = Wt + woff[6],  *w_iak = Wt + woff[7];
    const unsigned short* w_iav = Wt + woff[8],  *w_iao = Wt + woff[9];
    const unsigned short* w_tsq = Wt + woff[10], *w_tsk = Wt + woff[11];
    const unsigned short* w_tsv = Wt + woff[12], *w_tso = Wt + woff[13];
    const unsigned short* w_isq = Wt + woff[14], *w_isk = Wt + woff[15];
    const unsigned short* w_isv = Wt + woff[16], *w_iso = Wt + woff[17];

    // ---- projections ----
    gemm(te_bf, w_tp, b_tp, tp_bf, MT, DT, DD, stream);
    gemm(ie_bf, w_ip, b_ip, ip_bf, MI, DI, DD, stream);

    // ---- ta cross-attn: q=image_proj, k=text_proj, v=raw text ----
    gemm(ip_bf, w_taq, nullptr, Qb, MI, DD, DD, stream);
    gemm(tp_bf, w_tak, nullptr, Kb, MT, DD, DD, stream);
    gemm(te_bf, w_tav, nullptr, Vb, MT, DT, DD, stream);
    vtrans_kernel<<<(BB * HH * DH * LT) / 256, 256, 0, stream>>>(Vb, Vt, LT);
    attn_kernel<<<BB * HH, 128, 0, stream>>>(Qb, Kb, Vt, imask, tmask, Cx, LI, LT);
    gemm(Cx, w_tao, nullptr, ta_bf, MI, DD, DD, stream);

    // ---- ia cross-attn: q=text_proj, k=image_proj, v=raw image ----
    gemm(tp_bf, w_iaq, nullptr, Qb, MT, DD, DD, stream);
    gemm(ip_bf, w_iak, nullptr, Kb, MI, DD, DD, stream);
    gemm(ie_bf, w_iav, nullptr, Vb, MI, DI, DD, stream);
    vtrans_kernel<<<(BB * HH * DH * LI) / 256, 256, 0, stream>>>(Vb, Vt, LI);
    attn_kernel<<<BB * HH, 128, 0, stream>>>(Qb, Kb, Vt, tmask, imask, Cx, LT, LI);
    gemm(Cx, w_iao, nullptr, ia_bf, MT, DD, DD, stream);

    // ---- LayerNorm after cross-attn ----
    ln_kernel<<<MI, 256, 0, stream>>>(ta_bf, ln_g[0], ln_b[0], nullptr);
    ln_kernel<<<MT, 256, 0, stream>>>(ia_bf, ln_g[1], ln_b[1], nullptr);

    // ---- text self-attn (masks: image_mask for both; Lq=Lk=LI) ----
    gemm(ta_bf, w_tsq, nullptr, Qb, MI, DD, DD, stream);
    gemm(ta_bf, w_tsk, nullptr, Kb, MI, DD, DD, stream);
    gemm(ta_bf, w_tsv, nullptr, Vb, MI, DD, DD, stream);
    vtrans_kernel<<<(BB * HH * DH * LI) / 256, 256, 0, stream>>>(Vb, Vt, LI);
    attn_kernel<<<BB * HH, 128, 0, stream>>>(Qb, Kb, Vt, imask, imask, Cx, LI, LI);
    gemm(Cx, w_tso, nullptr, ta_bf, MI, DD, DD, stream);

    // ---- image self-attn (masks: text_mask for both; Lq=Lk=LT) ----
    gemm(ia_bf, w_isq, nullptr, Qb, MT, DD, DD, stream);
    gemm(ia_bf, w_isk, nullptr, Kb, MT, DD, DD, stream);
    gemm(ia_bf, w_isv, nullptr, Vb, MT, DD, DD, stream);
    vtrans_kernel<<<(BB * HH * DH * LT) / 256, 256, 0, stream>>>(Vb, Vt, LT);
    attn_kernel<<<BB * HH, 128, 0, stream>>>(Qb, Kb, Vt, tmask, tmask, Cx, LT, LT);
    gemm(Cx, w_iso, nullptr, ia_bf, MT, DD, DD, stream);

    // ---- final LayerNorms -> fp32 outputs (text first, then image) ----
    float* out = (float*)d_out;
    ln_kernel<<<MI, 256, 0, stream>>>(ta_bf, ln_g[2], ln_b[2], out);
    ln_kernel<<<MT, 256, 0, stream>>>(ia_bf, ln_g[3], ln_b[3], out + (size_t)MI * DD);
}
